// EdgeEncoder_1365799600695
// MI455X (gfx1250) — matrix operand order repaired and checked by reading, not compiled, not run on hardware
//
#include <hip/hip_runtime.h>
#include <math.h>

typedef _Float16 f16;
typedef __attribute__((ext_vector_type(16))) _Float16 v16h;
typedef __attribute__((ext_vector_type(8)))  float    v8f;
typedef __attribute__((ext_vector_type(4)))  _Float16 h4;

#define C_DIM  256
#define H1_DIM 128
#define H2_DIM 64
#define BN_EPS 1e-5f

// LDS row pads (halves) to break 512B-stride bank conflicts
#define XP   264   // x tile rows (256 + 8); h (128 ch) aliases first halves of own rows
#define W1P  264   // W1' rows (256 + 8)
#define W2P  136   // W2 rows (128 + 8)
#define TILE_E 128

#if defined(__has_builtin)
#if __has_builtin(__builtin_amdgcn_sched_group_barrier)
#define SGB(mask, size, id) __builtin_amdgcn_sched_group_barrier(mask, size, id)
#endif
#endif
#ifndef SGB
#define SGB(mask, size, id)
#endif

// ---------------- workspace layout (bytes) ----------------
// [0,     2048) : stats: sum[256], sumsq[256]  (f32)
// [2048,  2560) : b1' f32[128]
// [2560, 68096) : W1' f16[128*256]
// [68096,84480) : W2  f16[64*128]

__global__ void k_zero(float* stats) {
    int i = threadIdx.x;
    if (i < 2 * C_DIM) stats[i] = 0.0f;
}

// Per-channel sum / sum-of-squares of x = [avg(node[src],node[dst]) | edge_feat]
__global__ void k_stats(const float* __restrict__ node,
                        const float* __restrict__ ef,
                        const long long* __restrict__ ei,
                        float* __restrict__ stats, int E) {
    const int tid = threadIdx.x;           // tid == channel index (0..255)
    float s = 0.0f, sq = 0.0f;
    for (int e = blockIdx.x; e < E; e += gridDim.x) {
        float v;
        if (tid < 128) {
            int src = (int)ei[e];
            int dst = (int)ei[E + e];
            v = 0.5f * (node[src * 128 + tid] + node[dst * 128 + tid]);
        } else {
            v = ef[(long long)e * 128 + (tid - 128)];
        }
        s += v; sq += v * v;
    }
    atomicAdd(&stats[tid], s);
    atomicAdd(&stats[C_DIM + tid], sq);
}

// Fold BN into W1/b1, convert weights to f16
__global__ void k_fold(const float* __restrict__ W1, const float* __restrict__ b1,
                       const float* __restrict__ W2,
                       const float* __restrict__ gamma, const float* __restrict__ beta,
                       const float* __restrict__ stats, float invE,
                       f16* __restrict__ w1f, float* __restrict__ b1f,
                       f16* __restrict__ w2f) {
    __shared__ float s_scale[C_DIM], s_shift[C_DIM];
    const int tid = threadIdx.x;           // 256 threads, 1 block
    float mean = stats[tid] * invE;
    float var  = stats[C_DIM + tid] * invE - mean * mean;
    float sc   = gamma[tid] * rsqrtf(var + BN_EPS);
    s_scale[tid] = sc;
    s_shift[tid] = beta[tid] - mean * sc;
    __syncthreads();
    for (int i = tid; i < H1_DIM * C_DIM; i += 256)
        w1f[i] = (f16)(W1[i] * s_scale[i & 255]);
    if (tid < H1_DIM) {
        float a = b1[tid];
        for (int c = 0; c < C_DIM; ++c) a += W1[tid * C_DIM + c] * s_shift[c];
        b1f[tid] = a;
    }
    for (int i = tid; i < H2_DIM * H1_DIM; i += 256)
        w2f[i] = (f16)W2[i];
}

// Branch-free tanh-form GELU using HW transcendentals (v_exp_f32 / v_rcp_f32).
// 0.5*(1+tanh(u)) = 1 - 1/(e^{2u}+1);  2u*log2(e) = v*(2.3021694 + 0.1029415*v^2)
__device__ __forceinline__ float fast_gelu(float v) {
    float p = v * v;
    float z = v * (2.3021694f + 0.10294150f * p);
    float a = __builtin_amdgcn_exp2f(z);
    float r = __builtin_amdgcn_rcpf(a + 1.0f);
    return v * (1.0f - r);
}

// Fused: gather/avg/concat -> GEMM1(f16 WMMA) -> bias+GELU -> GEMM2 -> +b2 -> out
__global__ __launch_bounds__(256) void k_main(
    const float* __restrict__ node, const float* __restrict__ ef,
    const long long* __restrict__ ei,
    const f16* __restrict__ w1f, const float* __restrict__ b1f,
    const f16* __restrict__ w2f, const float* __restrict__ b2,
    float* __restrict__ out, int E) {
    extern __shared__ f16 smem[];
    f16* xs  = smem;                         // 128 x XP  (x, later aliased by h)
    f16* w1s = xs  + TILE_E * XP;            // 128 x W1P
    f16* w2s = w1s + H1_DIM * W1P;           // 64  x W2P

    const int tid = threadIdx.x;
    const int e0  = blockIdx.x * TILE_E;

    // ---- stage W1' and W2 into LDS (8-byte chunks, cooperative) ----
    const unsigned long long* w1g = (const unsigned long long*)w1f;
    for (int i = tid; i < H1_DIM * C_DIM / 4; i += 256) {
        int h = i >> 6, c = (i & 63) << 2;       // row = 64 ull
        *(unsigned long long*)(w1s + h * W1P + c) = w1g[i];
    }
    const unsigned long long* w2g = (const unsigned long long*)w2f;
    for (int i = tid; i < H2_DIM * H1_DIM / 4; i += 256) {
        int h = i >> 5, c = (i & 31) << 2;       // row = 32 ull
        *(unsigned long long*)(w2s + h * W2P + c) = w2g[i];
    }

    // ---- stage x tile (wave-private: wave w writes exactly rows 16w..16w+15) ----
    {
        int row = tid >> 1, half = tid & 1;
        int e = e0 + row;
        f16* dst = xs + row * XP + half * 128;
        if (e < E) {
            if (half == 0) {
                long long s = ei[e], d = ei[E + e];
                const float4* a4 = (const float4*)(node + s * 128);
                const float4* b4 = (const float4*)(node + d * 128);
                #pragma unroll
                for (int c = 0; c < 32; ++c) {
                    float4 a = a4[c], b = b4[c];
                    h4 hv;
                    hv.x = (f16)(0.5f * (a.x + b.x)); hv.y = (f16)(0.5f * (a.y + b.y));
                    hv.z = (f16)(0.5f * (a.z + b.z)); hv.w = (f16)(0.5f * (a.w + b.w));
                    *(h4*)(dst + 4 * c) = hv;
                }
            } else {
                const float4* f4 = (const float4*)(ef + (long long)e * 128);
                #pragma unroll
                for (int c = 0; c < 32; ++c) {
                    float4 a = f4[c];
                    h4 hv;
                    hv.x = (f16)a.x; hv.y = (f16)a.y; hv.z = (f16)a.z; hv.w = (f16)a.w;
                    *(h4*)(dst + 4 * c) = hv;
                }
            }
        } else {
            h4 z = {};
            #pragma unroll
            for (int c = 0; c < 32; ++c) *(h4*)(dst + 4 * c) = z;
        }
    }
    __syncthreads();   // for cooperative W1'/W2 staging

    const int lane = tid & 31;
    const int wid  = tid >> 5;
    const int m0   = wid * 16;          // 16 edges per wave
    const int lrow = lane & 15;
    const int lhi  = lane >> 4;         // 0 | 1
    const int kb   = lhi * 16;          // A-fragment K base offset (contiguous, matches B)

    // ---- layer 1: [16x256] @ W1'^T -> 8 accumulators of 16x16 f32 ----
    v8f acc[8] = {};
    const f16* arow = xs + (m0 + lrow) * XP;
    #pragma unroll
    for (int ks = 0; ks < 8; ++ks) {
        const int kk = ks * 32;
        v16h a;
        ((uint4*)&a)[0] = *(const uint4*)(arow + kk + kb);        // K kk+kb..+7
        ((uint4*)&a)[1] = *(const uint4*)(arow + kk + kb + 8);    // K kk+kb+8..+15
        v16h bf[8];
        #pragma unroll
        for (int nt = 0; nt < 8; ++nt) {
            const f16* bp = w1s + (nt * 16 + lrow) * W1P + kk + lhi * 16;
            ((uint4*)&bf[nt])[0] = *(const uint4*)(bp);
            ((uint4*)&bf[nt])[1] = *(const uint4*)(bp + 8);
        }
        #pragma unroll
        for (int nt = 0; nt < 8; ++nt)
            acc[nt] = __builtin_amdgcn_wmma_f32_16x16x32_f16(
                false, a, false, bf[nt], (short)0, acc[nt], false, false);
        // Pipeline directive: all 18 DS reads of this k-step scheduled before
        // its 8 WMMAs -> 8 live B fragments, partial dscnt waits, load/compute
        // overlap across k-steps.
        SGB(0x100, 18, 0);   // DS read group
        SGB(0x008, 8, 0);    // MFMA/WMMA group
    }

    // ---- bias + fast GELU -> h aliases first 128 halves of this wave's x rows ----
    // Safe: all layer-1 x reads by this wave precede these writes in program
    // order, and LDS ops from one wave complete in order.
    #pragma unroll
    for (int nt = 0; nt < 8; ++nt) {
        const int n = nt * 16 + lrow;
        const float bias = b1f[n];
        #pragma unroll
        for (int r = 0; r < 8; ++r) {
            float g = fast_gelu(acc[nt][r] + bias);
            xs[(m0 + r + 8 * lhi) * XP + n] = (f16)g;   // h(M=r+8*lhi, N=n)
        }
    }

    // ---- layer 2: [16x128] @ W2^T -> 4 accumulators ----
    v8f acc2[4] = {};
    const f16* arow2 = xs + (m0 + lrow) * XP;   // h rows, stride XP
    #pragma unroll
    for (int ks = 0; ks < 4; ++ks) {
        const int kk = ks * 32;
        v16h a;
        ((uint4*)&a)[0] = *(const uint4*)(arow2 + kk + kb);
        ((uint4*)&a)[1] = *(const uint4*)(arow2 + kk + kb + 8);
        v16h bf[4];
        #pragma unroll
        for (int nt = 0; nt < 4; ++nt) {
            const f16* bp = w2s + (nt * 16 + lrow) * W2P + kk + lhi * 16;
            ((uint4*)&bf[nt])[0] = *(const uint4*)(bp);
            ((uint4*)&bf[nt])[1] = *(const uint4*)(bp + 8);
        }
        #pragma unroll
        for (int nt = 0; nt < 4; ++nt)
            acc2[nt] = __builtin_amdgcn_wmma_f32_16x16x32_f16(
                false, a, false, bf[nt], (short)0, acc2[nt], false, false);
        SGB(0x100, 10, 0);   // DS read group
        SGB(0x008, 4, 0);    // MFMA/WMMA group
    }

    // ---- add b2, store [E,64] f32 ----
    #pragma unroll
    for (int nt = 0; nt < 4; ++nt) {
        const int n = nt * 16 + lrow;
        const float bias = b2[n];
        #pragma unroll
        for (int r = 0; r < 8; ++r) {
            const int e = e0 + m0 + r + 8 * lhi;
            if (e < E) out[(long long)e * H2_DIM + n] = acc2[nt][r] + bias;
        }
    }
}

extern "C" void kernel_launch(void* const* d_in, const int* in_sizes, int n_in,
                              void* d_out, int out_size, void* d_ws, size_t ws_size,
                              hipStream_t stream) {
    const float* node  = (const float*)d_in[0];
    const float* ef    = (const float*)d_in[1];
    const float* gamma = (const float*)d_in[2];
    const float* beta  = (const float*)d_in[3];
    const float* W1    = (const float*)d_in[4];
    const float* b1    = (const float*)d_in[5];
    const float* W2    = (const float*)d_in[6];
    const float* b2    = (const float*)d_in[7];
    const long long* ei = (const long long*)d_in[8];   // int64 [2,E]
    float* out = (float*)d_out;
    const int E = in_sizes[1] / 128;

    char* ws = (char*)d_ws;
    float* stats = (float*)ws;                 // 512 f32
    float* b1f   = (float*)(ws + 2048);        // 128 f32
    f16*   w1f   = (f16*)(ws + 2560);          // 32768 f16
    f16*   w2f   = (f16*)(ws + 68096);         // 8192 f16

    k_zero<<<1, 512, 0, stream>>>(stats);
    k_stats<<<2048, 256, 0, stream>>>(node, ef, ei, stats, E);
    k_fold<<<1, 256, 0, stream>>>(W1, b1, W2, gamma, beta, stats,
                                  1.0f / (float)E, w1f, b1f, w2f);
    const size_t smem =
        (size_t)(TILE_E * XP + H1_DIM * W1P + H2_DIM * W2P) * sizeof(f16);  // 149 KB
    k_main<<<(E + TILE_E - 1) / TILE_E, 256, smem, stream>>>(
        node, ef, ei, w1f, b1f, w2f, b2, out, E);
}
